// LSTMNeuralHawkesProcess_4415226380289
// MI455X (gfx1250) — compile-verified, matrix operand order checked by reading, not run on hardware
//
#include <hip/hip_runtime.h>
#include <hip/hip_bf16.h>

#define HDIM   64
#define MAXLEN 1024
#define BATCH  64
#define TEND   100.0f
#define EPSF   1e-10f
#define NMC    30
#define NW     448   // 7*H
#define BPB    16    // batches per block (one 16-row M tile)

typedef __attribute__((ext_vector_type(16))) _Float16 v16h;
typedef __attribute__((ext_vector_type(8)))  _Float16 v8h;
typedef __attribute__((ext_vector_type(8)))  float    v8f;

__device__ __forceinline__ float sigmoidf_(float x) { return 1.f / (1.f + __expf(-x)); }

// branchless, numerically stable: softplus(x) = max(x,0) + log1p(exp(-|x|))
__device__ __forceinline__ float softplusf_(float x) {
    return fmaxf(x, 0.f) + __logf(1.f + __expf(-fabsf(x)));
}

// branchless tanh via exp of the negative magnitude
__device__ __forceinline__ float tanhf_(float x) {
    float e = __expf(-2.f * fabsf(x));
    float t = (1.f - e) / (1.f + e);
    return copysignf(t, x);
}

// dt(i) exactly as the reference's _delta_t builds it (incl. -1 padding)
__device__ __forceinline__ float dt_at(const float* __restrict__ pads, int len, int j) {
    if (j < len)  return (j == 0) ? pads[0] : (pads[j] - pads[j - 1]);
    if (j == len) return TEND - pads[len - 1];
    return -1.f;
}

__device__ __forceinline__ float rnd01(unsigned s, unsigned b, unsigned k) {
    unsigned x = s * 747796405u ^ (b * 2891336453u) ^ (k * 1597334677u);
    x = x * 747796405u + 2891336453u;
    x ^= x >> 16; x *= 2246822519u; x ^= x >> 13;
    return (float)(x >> 8) * (1.0f / 16777216.0f);
}

__global__ __launch_bounds__(1024)
void hawkes_ctlstm_kernel(const float* __restrict__ seq_pads,
                          const float* __restrict__ W_rec,
                          const float* __restrict__ b_rec_g,
                          const float* __restrict__ W_f,
                          const float* __restrict__ b_f_g,
                          const int*   __restrict__ seq_lens,
                          float* __restrict__ block_partial)
{
    extern __shared__ char smem[];
    float* v_lds  = (float*)smem;             // [16][448] gate pre-activations (h@W)
    float* cstate = v_lds  + BPB * NW;        // [16][64]  c (decayed carry)
    float* cbarst = cstate + BPB * HDIM;      // [16][64]  cbar
    float* w0     = cbarst + BPB * HDIM;      // [448] W_rec row 0 (dt input row)
    float* brec   = w0     + NW;              // [448]
    float* wf     = brec   + NW;              // [64]
    float* dtin   = wf     + HDIM;            // [16]
    float* dtout  = dtin   + BPB;             // [16]
    float* evpart = dtout  + BPB;             // [16][2]  event-dot partials
    float* mcpart = evpart + BPB * 2;         // [16][30][2] MC-dot partials
    float* ll     = mcpart + BPB * NMC * 2;   // [16] per-batch loglike
    _Float16* h_lds = (_Float16*)(ll + BPB);  // [16][64] f16 hidden (WMMA A source)

    const int t    = threadIdx.x;
    const int lane = t & 31;
    const int wave = t >> 5;
    const float bf = b_f_g[0];
    const int b0   = blockIdx.x * BPB;

    // ---- setup: stage constants, zero recurrent state ----
    for (int i = t; i < NW; i += 1024) { w0[i] = W_rec[i]; brec[i] = b_rec_g[i]; }
    if (t < HDIM) wf[t] = W_f[t];
    cstate[t] = 0.f; cbarst[t] = 0.f; h_lds[t] = (_Float16)0.f;
    if (t < BPB) {
        const int bg  = b0 + t;
        const float* prow = seq_pads + (size_t)bg * MAXLEN;
        const float l0 = softplusf_(bf) + EPSF;   // intensity at zero init state
        ll[t] = __logf(l0) - l0 * prow[0];        // event k=0 term - MC k=0 term
    }

    // ---- resident WMMA B fragments: W_rec rows 1..64 -> f16, wave w = N-tile w ----
    v16h bfr0 = {}, bfr1 = {};
    const int n0 = wave * 16;
    if (wave < 28) {
        const int n = n0 + (lane & 15);
        const int khbase = (lane >> 4) * 8;     // lanes 16-31 hold K+8 / K+24
#pragma unroll
        for (int e = 0; e < 16; ++e) {
            const int k = (e & 7) + ((e >> 3) << 4) + khbase;
            bfr0[e] = (_Float16)W_rec[(size_t)(1 + k)      * NW + n];
            bfr1[e] = (_Float16)W_rec[(size_t)(1 + k + 32) * NW + n];
        }
    }
    __syncthreads();

    const int bloc  = t >> 6;        // local batch 0..15
    const int hh    = t & 63;        // hidden unit
    const int hhalf = (t >> 5) & 1;  // which wave-half of this batch's 64 lanes
    const int bg_t  = b0 + bloc;

    for (int j = 0; j < MAXLEN; ++j) {
        // ---- phase 0: per-batch delta-t for this scan step ----
        if (t < BPB) {
            const int bg  = b0 + t;
            const int len = seq_lens[bg];
            const float* prow = seq_pads + (size_t)bg * MAXLEN;
            __builtin_prefetch(prow + j + 2, 0, 1);   // global_prefetch_b8
            dtin[t]  = dt_at(prow, len, j);
            dtout[t] = dt_at(prow, len, j + 1);
        }
        // ---- phase 1: v = h @ W_rec[1:] via v_wmma_f32_16x16x32_f16 ----
        if (wave < 28) {
            const int row   = lane & 15;
            const int khalf = (lane >> 4) * 8;
            const _Float16* hrow = h_lds + row * HDIM;
            v8h a0lo = *(const v8h*)(hrow + khalf);
            v8h a0hi = *(const v8h*)(hrow + khalf + 16);
            v8h a1lo = *(const v8h*)(hrow + 32 + khalf);
            v8h a1hi = *(const v8h*)(hrow + 32 + khalf + 16);
            v16h a0, a1;
#pragma unroll
            for (int e = 0; e < 8; ++e) {
                a0[e] = a0lo[e]; a0[e + 8] = a0hi[e];
                a1[e] = a1lo[e]; a1[e + 8] = a1hi[e];
            }
            v8f acc = {0.f, 0.f, 0.f, 0.f, 0.f, 0.f, 0.f, 0.f};
            acc = __builtin_amdgcn_wmma_f32_16x16x32_f16(false, a0, false, bfr0,
                                                         (short)0, acc, false, false);
            acc = __builtin_amdgcn_wmma_f32_16x16x32_f16(false, a1, false, bfr1,
                                                         (short)0, acc, false, false);
            const int n  = n0 + (lane & 15);
            const int mb = (lane >> 4) * 8;
#pragma unroll
            for (int r = 0; r < 8; ++r) v_lds[(mb + r) * NW + n] = acc[r];
        }
        __syncthreads();

        // ---- phase 2: gates + CT-LSTM state update + partial dots (all 1024 thr) ----
        {
            const float x   = dtin[bloc];
            const float dto = dtout[bloc];
            float vg[7];
#pragma unroll
            for (int g = 0; g < 7; ++g) {
                const int n = g * HDIM + hh;
                vg[g] = v_lds[bloc * NW + n] + x * w0[n] + brec[n];
            }
            const float gi  = sigmoidf_(vg[0]);
            const float gf  = sigmoidf_(vg[1]);
            const float gz  = tanhf_(vg[2]);
            const float go  = sigmoidf_(vg[3]);
            const float gib = sigmoidf_(vg[4]);
            const float gfb = sigmoidf_(vg[5]);
            const float dec = softplusf_(vg[6]);
            const float ct   = cstate[t];
            const float cbar = cbarst[t];
            const float c    = gf * ct + gi * gz;          // post-event cell
            const float cbn  = gfb * cbar + gib * gz;      // target cell
            const float cd   = c - cbn;
            const float cdec = cbn + cd * __expf(-dec * dto);
            const float before = go * tanhf_(cdec);        // hidden just before next event
            cstate[t] = cdec;
            cbarst[t] = cbn;
            h_lds[t]  = (_Float16)before;

            const float wfh = wf[hh];
            float ev = before * wfh;                       // event-intensity dot
#pragma unroll
            for (int off = 16; off >= 1; off >>= 1) ev += __shfl_xor(ev, off, 32);
            if (lane == 0) evpart[bloc * 2 + hhalf] = ev;

            const unsigned kk = (unsigned)(j + 1);         // Monte-Carlo integral samples
#pragma unroll 2
            for (int s = 0; s < NMC; ++s) {
                const float u  = rnd01((unsigned)s, (unsigned)bg_t, kk);
                const float hs = go * tanhf_(cbn + cd * __expf(-u * dto * dec));
                float p = hs * wfh;
#pragma unroll
                for (int off = 16; off >= 1; off >>= 1) p += __shfl_xor(p, off, 32);
                if (lane == 0) mcpart[(bloc * NMC + s) * 2 + hhalf] = p;
            }
        }
        __syncthreads();

        // ---- phase 3: per-batch loss accumulation (branchless masks) ----
        if (t < BPB) {
            const int k   = j + 1;
            const int len = seq_lens[b0 + t];
            const float lamb = softplusf_(evpart[t * 2] + evpart[t * 2 + 1] + bf) + EPSF;
            float integ = 0.f;
#pragma unroll 2
            for (int s = 0; s < NMC; ++s)
                integ += softplusf_(mcpart[(t * NMC + s) * 2] +
                                    mcpart[(t * NMC + s) * 2 + 1] + bf) + EPSF;
            integ = (integ * (1.f / (float)NMC)) * dtout[t];
            const float ev_m  = (k <  len) ? 1.f : 0.f;
            const float int_m = (k <= len) ? 1.f : 0.f;
            ll[t] += ev_m * __logf(lamb) - int_m * integ;
        }
        __syncthreads();
    }

    if (t == 0) {
        float s = 0.f;
#pragma unroll
        for (int b = 0; b < BPB; ++b) s += ll[b];
        block_partial[blockIdx.x] = s;
    }
}

__global__ void hawkes_reduce_kernel(const float* __restrict__ part,
                                     float* __restrict__ out) {
    if (threadIdx.x == 0 && blockIdx.x == 0)
        out[0] = -(part[0] + part[1] + part[2] + part[3]) * (1.f / (float)BATCH);
}

extern "C" void kernel_launch(void* const* d_in, const int* in_sizes, int n_in,
                              void* d_out, int out_size, void* d_ws, size_t ws_size,
                              hipStream_t stream) {
    const float* seq_pads = (const float*)d_in[0];
    const float* W_rec    = (const float*)d_in[1];
    const float* b_rec    = (const float*)d_in[2];
    const float* W_f      = (const float*)d_in[3];
    const float* b_f      = (const float*)d_in[4];
    const int*   lens     = (const int*)d_in[5];
    float* ws  = (float*)d_ws;
    float* out = (float*)d_out;

    const size_t smem_floats = (size_t)BPB * NW       // v_lds
                             + (size_t)BPB * HDIM * 2 // cstate + cbar
                             + (size_t)NW * 2         // w0 + brec
                             + HDIM                   // wf
                             + BPB * 2                // dtin + dtout
                             + BPB * 2                // evpart
                             + (size_t)BPB * NMC * 2  // mcpart
                             + BPB;                   // ll
    const size_t smem = smem_floats * sizeof(float) + (size_t)BPB * HDIM * 2; // + f16 h

    hawkes_ctlstm_kernel<<<dim3(BATCH / BPB), dim3(1024), smem, stream>>>(
        seq_pads, W_rec, b_rec, W_f, b_f, lens, ws);
    hawkes_reduce_kernel<<<dim3(1), dim3(32), 0, stream>>>(ws, out);
}
